// SparseMultiHeadAttention_48524540510840
// MI455X (gfx1250) — matrix operand order repaired and checked by reading, not compiled, not run on hardware
//
#include <hip/hip_runtime.h>
#include <hip/hip_bf16.h>
#include <math.h>
#include <stdint.h>

// Problem constants (from reference)
#define B_  1
#define TQ_ 256
#define TK_ 8192
#define C_  256
#define H_  4
#define D_  64
#define IH_ 8
#define ID_ 64
#define TOPK_ 2048
#define KD_ 256          // inner dim of every projection GEMM
#define WROW_ 132        // LDS row stride for W tile (128 data + 4 pad dwords)

typedef __attribute__((ext_vector_type(16))) __bf16 v16bf;
typedef __attribute__((ext_vector_type(8)))  float  v8f;
typedef unsigned int v4u __attribute__((ext_vector_type(4)));
typedef unsigned int v8u __attribute__((ext_vector_type(8)));

__device__ __forceinline__ __bf16 f2bf(float f) {
    unsigned u = __builtin_bit_cast(unsigned, f);
    unsigned short h = (unsigned short)(u >> 16);
    return __builtin_bit_cast(__bf16, h);
}

// 16-bit A 16x32 layout: lane L: M = L%16, group g = L/16.
// element e: K = (e/8)*16 + g*8 + (e%8)   (relative to k0)
__device__ __forceinline__ v16bf load_a_frag(const float* __restrict__ rowbase, int g) {
    const float* p = rowbase + g * 8;
    float4 a0 = *(const float4*)(p + 0);
    float4 a1 = *(const float4*)(p + 4);
    float4 a2 = *(const float4*)(p + 16);
    float4 a3 = *(const float4*)(p + 20);
    v16bf af;
    af[0]=f2bf(a0.x); af[1]=f2bf(a0.y); af[2]=f2bf(a0.z); af[3]=f2bf(a0.w);
    af[4]=f2bf(a1.x); af[5]=f2bf(a1.y); af[6]=f2bf(a1.z); af[7]=f2bf(a1.w);
    af[8]=f2bf(a2.x); af[9]=f2bf(a2.y); af[10]=f2bf(a2.z); af[11]=f2bf(a2.w);
    af[12]=f2bf(a3.x); af[13]=f2bf(a3.y); af[14]=f2bf(a3.z); af[15]=f2bf(a3.w);
    return af;
}

// 16-bit B 32x16 layout: lane L: N = L%16, group g = L/16.
// element e: K = (e/8)*16 + g*8 + (e%8)   (relative to k0)
__device__ __forceinline__ v16bf load_b_frag(const float* rowbase, int g) {
    const float* p = rowbase + g * 8;
    float4 b0 = *(const float4*)(p + 0);
    float4 b1 = *(const float4*)(p + 4);
    float4 b2 = *(const float4*)(p + 16);
    float4 b3 = *(const float4*)(p + 20);
    v16bf bf_;
    bf_[0]=f2bf(b0.x); bf_[1]=f2bf(b0.y); bf_[2]=f2bf(b0.z); bf_[3]=f2bf(b0.w);
    bf_[4]=f2bf(b1.x); bf_[5]=f2bf(b1.y); bf_[6]=f2bf(b1.z); bf_[7]=f2bf(b1.w);
    bf_[8]=f2bf(b2.x); bf_[9]=f2bf(b2.y); bf_[10]=f2bf(b2.z); bf_[11]=f2bf(b2.w);
    bf_[12]=f2bf(b3.x); bf_[13]=f2bf(b3.y); bf_[14]=f2bf(b3.z); bf_[15]=f2bf(b3.w);
    return bf_;
}

// -----------------------------------------------------------------------------
// TDM: DMA one 2D W tile (64 rows x 128 floats, row stride KD_ floats in
// memory) into LDS, with TDM padding of 4 dwords every 128 dwords so the LDS
// row stride becomes WROW_=132 floats (bank-conflict-free fragment reads).
// Descriptor per CDNA5 ISA D# groups 0/1; groups 2/3 zero
// (tile_dim2/3/4 = 0 => unused dims).
// -----------------------------------------------------------------------------
__device__ __forceinline__ void tdm_load_w_tile(const float* gsrc, unsigned lds_off,
                                                unsigned Nrows) {
    unsigned long long ga = (unsigned long long)(uintptr_t)gsrc;
    v4u g0; v8u g1; v4u gz = {0u, 0u, 0u, 0u};
    g0[0] = 1u;                                           // count=1 (valid user D#)
    g0[1] = lds_off;                                      // lds_addr
    g0[2] = (unsigned)ga;                                 // global_addr[31:0]
    g0[3] = (unsigned)((ga >> 32) & 0x1FFFFFFu) | (2u << 30); // addr[56:32] | type=2
    // data_size=4B(code 2), pad_enable, pad_interval=128dw(code 6), pad_amount=4dw(code 3)
    g1[0] = (2u << 16) | (1u << 20) | (6u << 22) | (3u << 25);
    g1[1] = ((unsigned)KD_ & 0xFFFFu) << 16;              // tensor_dim0 lo16
    g1[2] = (((unsigned)KD_ >> 16) & 0xFFFFu) | ((Nrows & 0xFFFFu) << 16); // dim0 hi | dim1 lo
    g1[3] = ((Nrows >> 16) & 0xFFFFu) | (128u << 16);     // dim1 hi | tile_dim0=128
    g1[4] = 64u;                                          // tile_dim1=64, tile_dim2=0
    g1[5] = (unsigned)KD_;                                // tensor_dim0_stride lo32
    g1[6] = 0u;                                           // stride hi | dim1_stride lo
    g1[7] = 0u;
    asm volatile("tensor_load_to_lds %0, %1, %2, %2"
                 :: "s"(g0), "s"(g1), "s"(gz)
                 : "memory");
}

// -----------------------------------------------------------------------------
// Y[m,n] = sum_k X[m,k] * W[n,k] + bias[n]   (Y = X @ W.T + b), Kd fixed = 256.
// Block: 256 threads = 8 waves, covers 128 M x 64 N.
// Per wave: 16 M rows x 4 accumulator tiles (64 N cols) -> A fragment reused 4x.
// W tile staged in LDS by the Tensor Data Mover (one producer wave), K split
// into two 128-chunks; consumers read bf16 B fragments from padded LDS rows.
// -----------------------------------------------------------------------------
__global__ void gemm_xwT_wmma(const float* __restrict__ X,
                              const float* __restrict__ W,
                              const float* __restrict__ bias,
                              float* __restrict__ Y,
                              int M, int N) {
    __shared__ float lds_w[64 * WROW_];   // 33,792 bytes
    const int lane = threadIdx.x & 31;
    const int wave = threadIdx.x >> 5;
    const int g    = lane >> 4;
    const int lm   = lane & 15;
    const int m0   = blockIdx.y * 128 + wave * 16;
    const int n0   = blockIdx.x * 64;
    (void)M;

    const unsigned lds_base = (unsigned)(uintptr_t)&lds_w[0];
    const float* arow = X + (size_t)(m0 + lm) * KD_;

    v8f acc[4];
    #pragma unroll
    for (int c = 0; c < 4; ++c) acc[c] = (v8f){};

    #pragma unroll
    for (int half = 0; half < 2; ++half) {
        const int kbase = half * 128;
        if (threadIdx.x < 32) {                         // producer wave: issue TDM
            tdm_load_w_tile(W + (size_t)n0 * KD_ + kbase, lds_base, (unsigned)N);
            __builtin_amdgcn_s_wait_tensorcnt(0);
        }
        __syncthreads();

        __builtin_prefetch(arow + kbase + 128, 0, 1);   // prefetch next A chunk

        #pragma unroll
        for (int kk = 0; kk < 128; kk += 32) {
            v16bf af = load_a_frag(arow + kbase + kk, g);
            #pragma unroll
            for (int c = 0; c < 4; ++c) {
                const float* brow = lds_w + (size_t)(c * 16 + lm) * WROW_ + kk;
                v16bf bf_ = load_b_frag(brow, g);
                acc[c] = __builtin_amdgcn_wmma_f32_16x16x32_bf16(
                             false, af, false, bf_, (short)0, acc[c], false, false);
            }
        }
        __syncthreads();                                // before TDM overwrites tile
    }

    #pragma unroll
    for (int c = 0; c < 4; ++c) {
        const int col = n0 + c * 16 + lm;
        float bv = bias ? bias[col] : 0.0f;
        #pragma unroll
        for (int r = 0; r < 8; ++r) {
            int row = m0 + g * 8 + r;   // C/D layout: lanes 0-15 -> M=r, 16-31 -> M=r+8
            Y[(size_t)row * N + col] = acc[c][r] + bv;
        }
    }
}

// -----------------------------------------------------------------------------
// scores_idx[q,k] = sum_h softmax(hw)[h] * relu( (1/8) * dot(iq[q,h,:], ik[k,h,:]) )
// iq: (TQ, IH*ID) fp32, ik: (TK, IH*ID) fp32. One 16q x 16k tile per wave;
// 8 heads x 2 WMMA k-steps fully unrolled.
// -----------------------------------------------------------------------------
__global__ void index_score_wmma(const float* __restrict__ iq,
                                 const float* __restrict__ ik,
                                 const float* __restrict__ hw,
                                 float* __restrict__ scores) {
    const int lane = threadIdx.x & 31;
    const int wave = threadIdx.x >> 5;
    const int g    = lane >> 4;
    const int lm   = lane & 15;
    const int q0   = blockIdx.y * 16;
    const int k0   = blockIdx.x * 128 + wave * 16;

    // softmax over the 8 head weights (tiny; redundant per thread)
    float wv[IH_];
    float mx = -1e30f;
    #pragma unroll
    for (int h = 0; h < IH_; ++h) { wv[h] = hw[h]; mx = fmaxf(mx, wv[h]); }
    float sm = 0.0f;
    #pragma unroll
    for (int h = 0; h < IH_; ++h) { wv[h] = __expf(wv[h] - mx); sm += wv[h]; }
    float inv = 1.0f / sm;
    #pragma unroll
    for (int h = 0; h < IH_; ++h) wv[h] *= inv;

    const float scale = 0.125f;  // 1/sqrt(ID)
    const int Kd = IH_ * ID_;    // 512
    const float* arow = iq + (size_t)(q0 + lm) * Kd;
    const float* brow = ik + (size_t)(k0 + lm) * Kd;

    float tot[8];
    #pragma unroll
    for (int r = 0; r < 8; ++r) tot[r] = 0.0f;

    #pragma unroll
    for (int h = 0; h < IH_; ++h) {
        v8f acc = {};
        #pragma unroll
        for (int kk = 0; kk < ID_; kk += 32) {
            v16bf af = load_a_frag(arow + h * ID_ + kk, g);
            v16bf bf_ = load_b_frag(brow + h * ID_ + kk, g);
            acc = __builtin_amdgcn_wmma_f32_16x16x32_bf16(
                      false, af, false, bf_, (short)0, acc, false, false);
        }
        #pragma unroll
        for (int r = 0; r < 8; ++r) {
            float s = acc[r] * scale;
            s = s > 0.0f ? s : 0.0f;        // relu
            tot[r] += wv[h] * s;
        }
    }

    #pragma unroll
    for (int r = 0; r < 8; ++r) {
        int qrow = q0 + g * 8 + r;
        scores[(size_t)qrow * TK_ + (k0 + lm)] = tot[r];
    }
}

// -----------------------------------------------------------------------------
// Deterministic per-query top-K selection (unordered set; attention result is
// permutation invariant). One block (256 threads) per query. Threshold by
// bisection (LDS atomic counts), then index-ordered scan compaction.
// -----------------------------------------------------------------------------
__global__ void topk_kernel(const float* __restrict__ scores,
                            int* __restrict__ topk) {
    const int q = blockIdx.x;
    const int t = threadIdx.x;
    const float* row = scores + (size_t)q * TK_;
    int* out = topk + (size_t)q * TOPK_;

    __shared__ float redf[256];
    __shared__ int   redi[256];
    __shared__ int   cnt;
    __shared__ float s_lo, s_hi;

    const int CH = TK_ / 256;          // 32 contiguous elements per thread
    const int base = t * CH;

    // block max (scores are >= 0 due to relu)
    float mx = 0.0f;
    for (int j = 0; j < CH; ++j) mx = fmaxf(mx, row[base + j]);
    redf[t] = mx; __syncthreads();
    for (int s = 128; s > 0; s >>= 1) {
        if (t < s) redf[t] = fmaxf(redf[t], redf[t + s]);
        __syncthreads();
    }
    if (t == 0) { s_lo = 0.0f; s_hi = redf[0] + 1e-6f; }
    __syncthreads();

    // bisection: invariant count(> s_hi) <= TOPK
    for (int it = 0; it < 24; ++it) {
        float mid = 0.5f * (s_lo + s_hi);
        if (t == 0) cnt = 0;
        __syncthreads();
        int c = 0;
        for (int j = 0; j < CH; ++j) c += (row[base + j] > mid) ? 1 : 0;
        atomicAdd(&cnt, c);
        __syncthreads();
        if (t == 0) { if (cnt > TOPK_) s_lo = mid; else s_hi = mid; }
        __syncthreads();
    }
    const float thr = s_hi;
    __syncthreads();

    // pass 1: strictly above threshold (count <= TOPK). scan compaction.
    int c1 = 0;
    for (int j = 0; j < CH; ++j) c1 += (row[base + j] > thr) ? 1 : 0;
    redi[t] = c1; __syncthreads();
    for (int s = 1; s < 256; s <<= 1) {
        int v = (t >= s) ? redi[t - s] : 0;
        __syncthreads();
        redi[t] += v;
        __syncthreads();
    }
    int pos = redi[t] - c1;            // exclusive prefix
    int total1 = redi[255];
    for (int j = 0; j < CH; ++j) {
        if (row[base + j] > thr) { if (pos < TOPK_) out[pos] = base + j; ++pos; }
    }
    __syncthreads();

    // pass 2: everything else, index order, until full
    int c2 = 0;
    for (int j = 0; j < CH; ++j) c2 += (row[base + j] <= thr) ? 1 : 0;
    redi[t] = c2; __syncthreads();
    for (int s = 1; s < 256; s <<= 1) {
        int v = (t >= s) ? redi[t - s] : 0;
        __syncthreads();
        redi[t] += v;
        __syncthreads();
    }
    int pos2 = total1 + (redi[t] - c2);
    for (int j = 0; j < CH; ++j) {
        if (row[base + j] <= thr) { if (pos2 < TOPK_) out[pos2] = base + j; ++pos2; }
    }
}

// -----------------------------------------------------------------------------
// Gather attention: one block (256 threads) per query.
// Scores over selected keys, per-head softmax (LDS reductions), weighted V sum.
// -----------------------------------------------------------------------------
__global__ void gather_attention(const float* __restrict__ Q,
                                 const float* __restrict__ Kf,
                                 const float* __restrict__ Vf,
                                 const int* __restrict__ topk,
                                 float* __restrict__ out) {
    const int q = blockIdx.x;
    const int t = threadIdx.x;
    const int* ti = topk + (size_t)q * TOPK_;

    __shared__ float qrow[C_];             // 1 KB
    __shared__ float attn[H_ * TOPK_];     // 32 KB
    __shared__ float redf[256];
    __shared__ int   idxbuf[256];

    qrow[t] = Q[(size_t)q * C_ + t];
    __syncthreads();

    // phase 1: scores s[h, j] = (1/8) * dot(Q[h], K[idx_j, h])
    for (int j = t; j < TOPK_; j += 256) {
        int idx = ti[j];
        const float* krow = Kf + (size_t)idx * C_;
        #pragma unroll
        for (int h = 0; h < H_; ++h) {
            const float* kh = krow + h * D_;
            const float* qh = qrow + h * D_;
            float s = 0.0f;
            #pragma unroll 4
            for (int d = 0; d < D_; d += 4) {
                float4 kv = *(const float4*)(kh + d);
                s += kv.x * qh[d] + kv.y * qh[d + 1] + kv.z * qh[d + 2] + kv.w * qh[d + 3];
            }
            attn[h * TOPK_ + j] = s * 0.125f;   // 1/sqrt(D)
        }
    }
    __syncthreads();

    // phase 2: per-head softmax over TOPK
    for (int h = 0; h < H_; ++h) {
        float mx = -1e30f;
        for (int j = t; j < TOPK_; j += 256) mx = fmaxf(mx, attn[h * TOPK_ + j]);
        redf[t] = mx; __syncthreads();
        for (int s = 128; s > 0; s >>= 1) {
            if (t < s) redf[t] = fmaxf(redf[t], redf[t + s]);
            __syncthreads();
        }
        mx = redf[0]; __syncthreads();

        float sum = 0.0f;
        for (int j = t; j < TOPK_; j += 256) {
            float e = __expf(attn[h * TOPK_ + j] - mx);
            attn[h * TOPK_ + j] = e;
            sum += e;
        }
        redf[t] = sum; __syncthreads();
        for (int s = 128; s > 0; s >>= 1) {
            if (t < s) redf[t] += redf[t + s];
            __syncthreads();
        }
        float inv = 1.0f / redf[0]; __syncthreads();
        for (int j = t; j < TOPK_; j += 256) attn[h * TOPK_ + j] *= inv;
        __syncthreads();
    }

    // phase 3: out[h*64+d] = sum_j attn[h,j] * V[idx_j, h*64+d]
    const int h = t >> 6;
    const int d = t & 63;
    float acc = 0.0f;
    for (int c0 = 0; c0 < TOPK_; c0 += 256) {
        idxbuf[t] = ti[c0 + t];
        __syncthreads();
        #pragma unroll 8
        for (int j = 0; j < 256; ++j) {
            acc += attn[h * TOPK_ + c0 + j] *
                   Vf[(size_t)idxbuf[j] * C_ + h * D_ + d];
        }
        __syncthreads();
    }
    out[(size_t)q * C_ + t] = acc;
}

// -----------------------------------------------------------------------------
extern "C" void kernel_launch(void* const* d_in, const int* in_sizes, int n_in,
                              void* d_out, int out_size, void* d_ws, size_t ws_size,
                              hipStream_t stream) {
    const float* query = (const float*)d_in[0];   // (1, 256, 256)
    const float* key   = (const float*)d_in[1];   // (1, 8192, 256)
    // d_in[2] = value (unused by reference)
    const float* Wqkv  = (const float*)d_in[3];   // (768, 256)
    const float* bqkv  = (const float*)d_in[4];   // (768,)
    const float* Wout  = (const float*)d_in[5];   // (256, 256)
    const float* bout  = (const float*)d_in[6];   // (256,)
    const float* Wiq   = (const float*)d_in[7];   // (512, 256)
    const float* Wik   = (const float*)d_in[8];   // (512, 256)
    const float* hw    = (const float*)d_in[9];   // (8,)
    float* outp = (float*)d_out;                  // (1, 256, 256) fp32

    // workspace carve-up (fp32 unless noted)
    float* Kflat  = (float*)d_ws;                 // 8192*256
    float* Vflat  = Kflat  + (size_t)TK_ * C_;    // 8192*256
    float* Qbuf   = Vflat  + (size_t)TK_ * C_;    // 256*256
    float* iqb    = Qbuf   + (size_t)TQ_ * C_;    // 256*512
    float* ikb    = iqb    + (size_t)TQ_ * IH_ * ID_;  // 8192*512
    float* sidx   = ikb    + (size_t)TK_ * IH_ * ID_;  // 256*8192
    float* attout = sidx   + (size_t)TQ_ * TK_;   // 256*256
    int*   tkidx  = (int*)(attout + (size_t)TQ_ * C_); // 256*2048 ints

    dim3 blk(256);

    // K_flat = key @ Wqkv[C:2C].T + bqkv[C:2C]
    gemm_xwT_wmma<<<dim3(C_ / 64, TK_ / 128), blk, 0, stream>>>(
        key, Wqkv + (size_t)C_ * C_, bqkv + C_, Kflat, TK_, C_);
    // V_flat = key @ Wqkv[2C:3C].T + bqkv[2C:3C]
    gemm_xwT_wmma<<<dim3(C_ / 64, TK_ / 128), blk, 0, stream>>>(
        key, Wqkv + (size_t)2 * C_ * C_, bqkv + 2 * C_, Vflat, TK_, C_);
    // Q = query @ Wqkv[0:C].T + bqkv[0:C]
    gemm_xwT_wmma<<<dim3(C_ / 64, TQ_ / 128), blk, 0, stream>>>(
        query, Wqkv, bqkv, Qbuf, TQ_, C_);
    // iq = query @ Wiq.T
    gemm_xwT_wmma<<<dim3((IH_ * ID_) / 64, TQ_ / 128), blk, 0, stream>>>(
        query, Wiq, nullptr, iqb, TQ_, IH_ * ID_);
    // ik = key @ Wik.T
    gemm_xwT_wmma<<<dim3((IH_ * ID_) / 64, TK_ / 128), blk, 0, stream>>>(
        key, Wik, nullptr, ikb, TK_, IH_ * ID_);

    // index scores (relu + head-weight combine fused)
    index_score_wmma<<<dim3(TK_ / 128, TQ_ / 16), blk, 0, stream>>>(
        iqb, ikb, hw, sidx);

    // deterministic top-k per query
    topk_kernel<<<dim3(TQ_), blk, 0, stream>>>(sidx, tkidx);

    // gather attention
    gather_attention<<<dim3(TQ_), blk, 0, stream>>>(Qbuf, Kflat, Vflat, tkidx, attout);

    // final projection: out = attout @ Wout.T + bout
    gemm_xwT_wmma<<<dim3(C_ / 64, TQ_ / 128), blk, 0, stream>>>(
        attout, Wout, bout, outp, TQ_, C_);
}
